// GRUModel2_87806311400027
// MI455X (gfx1250) — matrix operand order repaired and checked by reading, not compiled, not run on hardware
//
#include <hip/hip_runtime.h>
#include <hip/hip_bf16.h>
#include <math.h>

// ---------------------------------------------------------------------------
// Problem constants (from reference): B=512, T=256, F=64, H1=256,H2=128,H3=64
// ---------------------------------------------------------------------------
constexpr int BATCH = 512;
constexpr int SEQ_T = 256;
constexpr int F_IN  = 64;
constexpr int H_1   = 256;
constexpr int H_2   = 128;
constexpr int H_3   = 64;

// f16 weight workspace layout (element offsets)
constexpr size_t OFF_IH1 = 0;
constexpr size_t OFF_HH1 = OFF_IH1 + (size_t)3 * H_1 * F_IN;   // 49152
constexpr size_t OFF_IH2 = OFF_HH1 + (size_t)3 * H_1 * H_1;    // 245760
constexpr size_t OFF_HH2 = OFF_IH2 + (size_t)3 * H_2 * H_1;    // 344064
constexpr size_t OFF_IH3 = OFF_HH2 + (size_t)3 * H_2 * H_2;    // 393216
constexpr size_t OFF_HH3 = OFF_IH3 + (size_t)3 * H_3 * H_2;    // 417792

// LDS pitches (halfs), padded by 8 so lanes hit distinct bank groups
constexpr int P_X  = F_IN + 8;   // 72
constexpr int P_H1 = H_1 + 8;    // 264
constexpr int P_H2 = H_2 + 8;    // 136
constexpr int P_H3 = H_3 + 8;    // 72

constexpr int SM_X  = 0;
constexpr int SM_H1 = SM_X  + 16 * P_X;
constexpr int SM_H2 = SM_H1 + 2 * 16 * P_H1;
constexpr int SM_H3 = SM_H2 + 2 * 16 * P_H2;
constexpr int SM_TOT = SM_H3 + 2 * 16 * P_H3;   // 16256 halfs = 32512 B

typedef __attribute__((ext_vector_type(16))) _Float16 v16h;
typedef __attribute__((ext_vector_type(8)))  _Float16 v8h;
typedef __attribute__((ext_vector_type(8)))  float    v8f;

union V16 { v16h v; v8h h[2]; };

// A fragment (16x32 f16, MxK) from row-major LDS tile, pitch in halfs.
// Lane<16: row=lane, K {0..7,16..23}; lane>=16: row=lane-16, K {8..15,24..31}.
__device__ __forceinline__ v16h ldA(const _Float16* base, int pitch, int k0, int lane) {
  int m = lane & 15, hs = lane >> 4;
  const _Float16* p = base + m * pitch + k0 + hs * 8;
  V16 r;
  r.h[0] = *(const v8h*)(p);
  r.h[1] = *(const v8h*)(p + 16);
  return r.v;
}

// B fragment (32x16 f16, KxN) from row-major weights W[row][k] (row = output col n).
// Lane<16: col n=lane, K {0..7,16..23}; lane>=16: col n=lane-16, K {8..15,24..31}.
__device__ __forceinline__ v16h ldB(const _Float16* __restrict__ W, int kw,
                                    int row0, int k0, int lane) {
  int n = lane & 15, hs = lane >> 4;
  const _Float16* p = W + (size_t)(row0 + n) * kw + k0 + hs * 8;
  V16 r;
  r.h[0] = *(const v8h*)(p);
  r.h[1] = *(const v8h*)(p + 16);
  return r.v;
}

__device__ __forceinline__ v8f wmma_f16(v16h a, v16h b, v8f c) {
  return __builtin_amdgcn_wmma_f32_16x16x32_f16(false, a, false, b, (short)0, c,
                                                false, false);
}

__device__ __forceinline__ float sigmf(float x) { return 1.0f / (1.0f + __expf(-x)); }

// One GRU column-tile (16 H-columns) for one 16-row batch tile, one timestep.
// KIN = input width, KH = hidden width of this layer.
template <int KIN, int KH>
__device__ __forceinline__ void gru_tile(
    const _Float16* aIn, int pIn,                  // LDS input activations [16][KIN]
    const _Float16* hPrev, int pH,                 // LDS previous h [16][KH]
    _Float16* hNext,                               // LDS next h [16][KH]
    const _Float16* __restrict__ Wih,              // global f16 [3*KH][KIN]
    const _Float16* __restrict__ Whh,              // global f16 [3*KH][KH]
    const float* __restrict__ bih, const float* __restrict__ bhh, // f32 [3*KH]
    int j, int lane)
{
  v8f axr = {0,0,0,0,0,0,0,0}, axz = {0,0,0,0,0,0,0,0}, axn = {0,0,0,0,0,0,0,0};
  v8f ahr = {0,0,0,0,0,0,0,0}, ahz = {0,0,0,0,0,0,0,0}, ahn = {0,0,0,0,0,0,0,0};
  const int nrow = j * 16;

  // input-side GEMM: [16,KIN] x [KIN,16] per gate
#pragma unroll
  for (int k0 = 0; k0 < KIN; k0 += 32) {
    v16h a  = ldA(aIn, pIn, k0, lane);
    v16h br = ldB(Wih, KIN, 0 * KH + nrow, k0, lane);
    v16h bz = ldB(Wih, KIN, 1 * KH + nrow, k0, lane);
    v16h bn = ldB(Wih, KIN, 2 * KH + nrow, k0, lane);
    axr = wmma_f16(a, br, axr);
    axz = wmma_f16(a, bz, axz);
    axn = wmma_f16(a, bn, axn);
  }
  // recurrent-side GEMM: [16,KH] x [KH,16] per gate
#pragma unroll
  for (int k0 = 0; k0 < KH; k0 += 32) {
    v16h a  = ldA(hPrev, pH, k0, lane);
    v16h br = ldB(Whh, KH, 0 * KH + nrow, k0, lane);
    v16h bz = ldB(Whh, KH, 1 * KH + nrow, k0, lane);
    v16h bn = ldB(Whh, KH, 2 * KH + nrow, k0, lane);
    ahr = wmma_f16(a, br, ahr);
    ahz = wmma_f16(a, bz, ahz);
    ahn = wmma_f16(a, bn, ahn);
  }

  // Elementwise gate math. D layout: lane holds column n, rows mbase..mbase+7.
  const int n   = lane & 15;
  const int col = nrow + n;
  const float bir = bih[0 * KH + col], biz = bih[1 * KH + col], bin = bih[2 * KH + col];
  const float bhr = bhh[0 * KH + col], bhz = bhh[1 * KH + col], bhn = bhh[2 * KH + col];
  const int mbase = (lane >> 4) * 8;
#pragma unroll
  for (int i = 0; i < 8; ++i) {
    float r  = sigmf(axr[i] + bir + ahr[i] + bhr);
    float z  = sigmf(axz[i] + biz + ahz[i] + bhz);
    float nn = tanhf(axn[i] + bin + r * (ahn[i] + bhn));
    float hp = (float)hPrev[(mbase + i) * pH + col];
    float hv = (1.0f - z) * nn + z * hp;
    hNext[(mbase + i) * pH + col] = (_Float16)hv;
  }
}

__global__ void f32_to_f16_kernel(const float* __restrict__ src,
                                  _Float16* __restrict__ dst, int n) {
  int i = blockIdx.x * blockDim.x + threadIdx.x;
  if (i < n) dst[i] = (_Float16)src[i];
}

// Persistent fused kernel: one workgroup per 16 batch rows, whole sequence,
// all three GRU layers + dense head. 8 waves of 32 (wave32).
__global__ void __launch_bounds__(256)
gru_fused_kernel(const float* __restrict__ x,
                 const _Float16* __restrict__ wf,
                 const float* __restrict__ bih1, const float* __restrict__ bhh1,
                 const float* __restrict__ bih2, const float* __restrict__ bhh2,
                 const float* __restrict__ bih3, const float* __restrict__ bhh3,
                 const float* __restrict__ Wd, const float* __restrict__ bd,
                 float* __restrict__ out)
{
  __shared__ __align__(16) _Float16 sm[SM_TOT];
  const int tid  = threadIdx.x;
  const int lane = tid & 31;
  const int wave = tid >> 5;
  const int b0   = blockIdx.x * 16;

  const _Float16* Wih1 = wf + OFF_IH1;
  const _Float16* Whh1 = wf + OFF_HH1;
  const _Float16* Wih2 = wf + OFF_IH2;
  const _Float16* Whh2 = wf + OFF_HH2;
  const _Float16* Wih3 = wf + OFF_IH3;
  const _Float16* Whh3 = wf + OFF_HH3;

  // zero LDS (h0 = 0)
  for (int i = tid; i < SM_TOT; i += 256) sm[i] = (_Float16)0.0f;
  __syncthreads();

  for (int t = 0; t < SEQ_T; ++t) {
    // stage x_t tile [16,64]: 256 threads x 4 floats, convert to f16
    {
      int idx = tid * 4;
      int row = idx >> 6, col = idx & 63;
      const float4 v = *(const float4*)(x + ((size_t)(b0 + row) * SEQ_T + t) * F_IN + col);
      _Float16* d = sm + SM_X + row * P_X + col;
      d[0] = (_Float16)v.x; d[1] = (_Float16)v.y;
      d[2] = (_Float16)v.z; d[3] = (_Float16)v.w;
    }
    __syncthreads();

    const int cur = t & 1, nxt = cur ^ 1;
    // ping/pong h-state bases computed via integer offsets (no LDS pointer
    // arrays -> avoids addrspacecast static initializers)
    _Float16* h1cur = sm + SM_H1 + cur * (16 * P_H1);
    _Float16* h1nxt = sm + SM_H1 + nxt * (16 * P_H1);
    _Float16* h2cur = sm + SM_H2 + cur * (16 * P_H2);
    _Float16* h2nxt = sm + SM_H2 + nxt * (16 * P_H2);
    _Float16* h3cur = sm + SM_H3 + cur * (16 * P_H3);
    _Float16* h3nxt = sm + SM_H3 + nxt * (16 * P_H3);

    // layer 1: 16 column tiles over 8 waves (2 each)
    for (int j = wave; j < H_1 / 16; j += 8)
      gru_tile<F_IN, H_1>(sm + SM_X, P_X, h1cur, P_H1, h1nxt,
                          Wih1, Whh1, bih1, bhh1, j, lane);
    __syncthreads();

    // layer 2: 8 column tiles, input = layer-1 output of this step
    for (int j = wave; j < H_2 / 16; j += 8)
      gru_tile<H_1, H_2>(h1nxt, P_H1, h2cur, P_H2, h2nxt,
                         Wih2, Whh2, bih2, bhh2, j, lane);
    __syncthreads();

    // layer 3: 4 column tiles
    for (int j = wave; j < H_3 / 16; j += 8)
      gru_tile<H_2, H_3>(h2nxt, P_H2, h3cur, P_H3, h3nxt,
                         Wih3, Whh3, bih3, bhh3, j, lane);
    __syncthreads();
  }

  // T=256 even -> final h3 lives in buffer 0. Dense: out[b] = h3[b] . Wd + bd
  if (tid < 16) {
    const _Float16* h = sm + SM_H3 + tid * P_H3;
    float acc = bd[0];
#pragma unroll
    for (int k = 0; k < H_3; ++k) acc += (float)h[k] * Wd[k];
    out[b0 + tid] = acc;
  }
}

extern "C" void kernel_launch(void* const* d_in, const int* in_sizes, int n_in,
                              void* d_out, int out_size, void* d_ws, size_t ws_size,
                              hipStream_t stream) {
  (void)in_sizes; (void)n_in; (void)out_size; (void)ws_size;
  const float* x     = (const float*)d_in[0];
  const float* Wih1  = (const float*)d_in[1];
  const float* Whh1  = (const float*)d_in[2];
  const float* bih1  = (const float*)d_in[3];
  const float* bhh1  = (const float*)d_in[4];
  const float* Wih2  = (const float*)d_in[5];
  const float* Whh2  = (const float*)d_in[6];
  const float* bih2  = (const float*)d_in[7];
  const float* bhh2  = (const float*)d_in[8];
  const float* Wih3  = (const float*)d_in[9];
  const float* Whh3  = (const float*)d_in[10];
  const float* bih3  = (const float*)d_in[11];
  const float* bhh3  = (const float*)d_in[12];
  const float* Wd    = (const float*)d_in[13];
  const float* bd    = (const float*)d_in[14];

  _Float16* wf = (_Float16*)d_ws;

  auto cvt = [&](const float* s, size_t off, int n) {
    f32_to_f16_kernel<<<(n + 255) / 256, 256, 0, stream>>>(s, wf + off, n);
  };
  cvt(Wih1, OFF_IH1, 3 * H_1 * F_IN);
  cvt(Whh1, OFF_HH1, 3 * H_1 * H_1);
  cvt(Wih2, OFF_IH2, 3 * H_2 * H_1);
  cvt(Whh2, OFF_HH2, 3 * H_2 * H_2);
  cvt(Wih3, OFF_IH3, 3 * H_3 * H_2);
  cvt(Whh3, OFF_HH3, 3 * H_3 * H_3);

  gru_fused_kernel<<<BATCH / 16, 256, 0, stream>>>(
      x, wf, bih1, bhh1, bih2, bhh2, bih3, bhh3, Wd, bd, (float*)d_out);
}
